// UnifiedPolicyValueNetwork_74844100100171
// MI455X (gfx1250) — matrix operand order repaired and checked by reading, not compiled, not run on hardware
//
#include <hip/hip_runtime.h>
#include <hip/hip_bf16.h>
#include <math.h>

// ---------------------------------------------------------------------------
// Problem constants (from reference)
// ---------------------------------------------------------------------------
#define NN 131072      // nodes
#define EE 1048576     // edges
#define FF 32          // node feature dim
#define HH 128         // hidden
#define GG 512         // graphs
#define NACT 5
#define MAXNEU 1000
#define NACTV 4

// output layout (flat concat, per reference return order)
#define OUT_A0   0                        // action_type   [512,5]
#define OUT_S0   (GG * NACT)              // source_logits [512,1000]
#define OUT_T0   (OUT_S0 + GG * MAXNEU)   // target_logits [512,1000]
#define OUT_ACT0 (OUT_T0 + GG * MAXNEU)   // activation    [512,4]
#define OUT_V0   (OUT_ACT0 + GG * NACTV)  // value         [512,1]

typedef __attribute__((ext_vector_type(16))) _Float16 v16h;
typedef __attribute__((ext_vector_type(8)))  float    v8f;

// ---------------------------------------------------------------------------
// WMMA helpers (wave32; one wave owns a 16x16 tile, K=32 per instruction)
// ---------------------------------------------------------------------------
__device__ __forceinline__ v8f wmma16(v16h a, v16h b, v8f c) {
  // v_wmma_f32_16x16x32_f16 : D = A(16x32 f16) * B(32x16 f16) + C(f32)
  return __builtin_amdgcn_wmma_f32_16x16x32_f16(
      /*neg_a=*/false, a, /*neg_b=*/false, b,
      /*c_mod=*/(short)0, c, /*reuse_a=*/false, /*reuse_b=*/false);
}

// Load a 16x32 A-tile from an fp32 row-major matrix, converting to f16,
// following the CDNA5 16-bit A layout:
//   lane l<16 : row M=m0+l,   halves 0..7 = K=k0..k0+7,  8..15 = K=k0+16..k0+23
//   lane l>=16: row M=m0+l-16, halves 0..7 = K=k0+8..+15, 8..15 = K=k0+24..+31
__device__ __forceinline__ v16h load_a_f32(const float* base, int m0, int k0, int ld) {
  const int lane = threadIdx.x & 31;
  const int m  = m0 + (lane & 15);
  const int kb = k0 + ((lane & 16) ? 8 : 0);
  const float* p = base + (size_t)m * ld + kb;
  v16h a;
#pragma unroll
  for (int j = 0; j < 8; ++j) a[j] = (_Float16)p[j];
#pragma unroll
  for (int j = 0; j < 8; ++j) a[j + 8] = (_Float16)p[16 + j];
  return a;
}

// B tiles are pre-packed so each lane does one contiguous 32-byte load.
__device__ __forceinline__ v16h load_bp(const _Float16* pack, int NT, int kt, int nt) {
  const int lane = threadIdx.x & 31;
  return *(const v16h*)(pack + (((size_t)(kt * NT + nt) * 32 + lane) << 4));
}

// ---------------------------------------------------------------------------
// Utility kernels
// ---------------------------------------------------------------------------
__global__ void k_zero(float* p, int n) {
  int i = blockIdx.x * blockDim.x + threadIdx.x;
  if (i < n) p[i] = 0.0f;
}

// Pack fp32 weight W[K,N] (row major) into WMMA B-matrix lane layout (f16).
// Packed element (kt,nt,lane,j):  n = nt*16 + (lane&15)
//                                 k = kt*32 + (lane<16 ? 0 : 8) + (j<8 ? j : j+8)
__global__ void k_pack_b(const float* __restrict__ W, _Float16* __restrict__ out,
                         int K, int N) {
  int i = blockIdx.x * blockDim.x + threadIdx.x;
  int NT = N >> 4;
  int total = (K >> 5) * NT * 512;
  if (i >= total) return;
  int j    = i & 15;
  int lane = (i >> 4) & 31;
  int tile = i >> 9;
  int nt = tile % NT;
  int kt = tile / NT;
  int n = nt * 16 + (lane & 15);
  int k = kt * 32 + ((lane & 16) ? 8 : 0) + (j < 8 ? j : j + 8);
  out[i] = (_Float16)W[(size_t)k * N + n];
}

// ---------------------------------------------------------------------------
// Stage 1: h0 = node_features @ W_in   ([N,32] x [32,128], single K-step WMMA)
// One wave per 16-row band; A loaded once, 8 column tiles.
// ---------------------------------------------------------------------------
__global__ __launch_bounds__(256) void k_h0(const float* __restrict__ nf,
                                            const _Float16* __restrict__ bpack,
                                            float* __restrict__ h0) {
  const int wave = threadIdx.x >> 5;
  const int lane = threadIdx.x & 31;
  const int m0 = (blockIdx.x * 8 + wave) * 16;
  v16h a = load_a_f32(nf, m0, 0, FF);
  const int col0  = (lane & 15);
  const int mbase = m0 + ((lane & 16) ? 8 : 0);
#pragma unroll
  for (int nt = 0; nt < 8; ++nt) {
    v8f c = {};
    c = wmma16(a, load_bp(bpack, 8, 0, nt), c);
    const int col = nt * 16 + col0;
#pragma unroll
    for (int r = 0; r < 8; ++r)
      h0[(size_t)(mbase + r) * HH + col] = c[r];
  }
}

// ---------------------------------------------------------------------------
// Stage 2: msg[tgt] += h0[src] * w    (consecutive lanes = consecutive feats)
// ---------------------------------------------------------------------------
__global__ void k_edges(const int* __restrict__ ei, const float* __restrict__ ew,
                        const float* __restrict__ h0, float* __restrict__ msg) {
  int i = blockIdx.x * blockDim.x + threadIdx.x;  // E*H threads
  int e = i >> 7;
  int f = i & 127;
  int s = ei[e];
  int t = ei[EE + e];
  float v = h0[(size_t)s * HH + f] * ew[e];
  atomicAdd(&msg[(size_t)t * HH + f], v);
}

// ---------------------------------------------------------------------------
// Stage 3: h = relu(h0 + msg @ W_msg), written IN PLACE over msg.
// One wave per 16-row band: all 4 A k-tiles in registers before any store.
// ---------------------------------------------------------------------------
__global__ __launch_bounds__(256) void k_msggemm(const float* __restrict__ h0,
                                                 float* __restrict__ hbuf,  // in: msg, out: h
                                                 const _Float16* __restrict__ bpack) {
  const int wave = threadIdx.x >> 5;
  const int lane = threadIdx.x & 31;
  const int m0 = (blockIdx.x * 8 + wave) * 16;
  v16h a0 = load_a_f32(hbuf, m0,  0, HH);
  v16h a1 = load_a_f32(hbuf, m0, 32, HH);
  v16h a2 = load_a_f32(hbuf, m0, 64, HH);
  v16h a3 = load_a_f32(hbuf, m0, 96, HH);
  const int col0  = (lane & 15);
  const int mbase = m0 + ((lane & 16) ? 8 : 0);
#pragma unroll
  for (int nt = 0; nt < 8; ++nt) {
    v8f c = {};
    c = wmma16(a0, load_bp(bpack, 8, 0, nt), c);
    c = wmma16(a1, load_bp(bpack, 8, 1, nt), c);
    c = wmma16(a2, load_bp(bpack, 8, 2, nt), c);
    c = wmma16(a3, load_bp(bpack, 8, 3, nt), c);
    const int col = nt * 16 + col0;
#pragma unroll
    for (int r = 0; r < 8; ++r) {
      size_t idx = (size_t)(mbase + r) * HH + col;
      float v = c[r] + h0[idx];
      hbuf[idx] = fmaxf(v, 0.0f);
    }
  }
}

// ---------------------------------------------------------------------------
// Stage 4: pooling. batch is sorted -> run-length accumulate, few atomics.
// ---------------------------------------------------------------------------
__global__ void k_cnt(const int* __restrict__ batch, float* __restrict__ cnt) {
  int i = blockIdx.x * blockDim.x + threadIdx.x;
  if (i < NN) atomicAdd(&cnt[batch[i]], 1.0f);
}

#define POOL_CHUNK 256
__global__ __launch_bounds__(128) void k_pool(const float* __restrict__ h,
                                              const int* __restrict__ batch,
                                              float* __restrict__ seg) {
  const int f  = threadIdx.x;               // 0..127 (one feature per lane)
  const int n0 = blockIdx.x * POOL_CHUNK;
  int gcur = batch[n0];
  float acc = 0.0f;
  for (int i = 0; i < POOL_CHUNK; ++i) {
    int node = n0 + i;
    int gb = batch[node];
    if (gb != gcur) {
      atomicAdd(&seg[(size_t)gcur * HH + f], acc);
      acc = 0.0f;
      gcur = gb;
    }
    acc += h[(size_t)node * HH + f];
  }
  atomicAdd(&seg[(size_t)gcur * HH + f], acc);
}

__global__ void k_mean(const float* __restrict__ seg, const float* __restrict__ cnt,
                       float* __restrict__ g) {
  int i = blockIdx.x * blockDim.x + threadIdx.x;  // G*H
  if (i >= GG * HH) return;
  float c = cnt[i >> 7];
  g[i] = (c > 0.0f) ? seg[i] / fmaxf(c, 1.0f) : 0.0f;
}

// ---------------------------------------------------------------------------
// Stage 5: backbone.  ge = [g,g] -> K=256 but both halves share A tiles.
// s1 = relu(ge @ W1) : [512,256]x[256,128]
// ---------------------------------------------------------------------------
__global__ __launch_bounds__(256) void k_s1(const float* __restrict__ g,
                                            const _Float16* __restrict__ bpack,
                                            float* __restrict__ s1) {
  const int wave = threadIdx.x >> 5;
  const int lane = threadIdx.x & 31;
  const int m0 = (blockIdx.x * 8 + wave) * 16;     // 32 bands of 16 rows
  v16h a[4];
#pragma unroll
  for (int kt = 0; kt < 4; ++kt) a[kt] = load_a_f32(g, m0, kt * 32, HH);
  const int col0  = (lane & 15);
  const int mbase = m0 + ((lane & 16) ? 8 : 0);
#pragma unroll
  for (int nt = 0; nt < 8; ++nt) {
    v8f c = {};
#pragma unroll
    for (int kk = 0; kk < 8; ++kk)              // ge has K=256; A repeats
      c = wmma16(a[kk & 3], load_bp(bpack, 8, kk, nt), c);
    const int col = nt * 16 + col0;
#pragma unroll
    for (int r = 0; r < 8; ++r)
      s1[(size_t)(mbase + r) * HH + col] = fmaxf(c[r], 0.0f);
  }
}

// s = relu(s1 @ W2) : [512,128]x[128,64]
__global__ __launch_bounds__(256) void k_s2(const float* __restrict__ s1,
                                            const _Float16* __restrict__ bpack,
                                            float* __restrict__ s) {
  const int wave = threadIdx.x >> 5;
  const int lane = threadIdx.x & 31;
  const int m0 = (blockIdx.x * 8 + wave) * 16;
  v16h a[4];
#pragma unroll
  for (int kt = 0; kt < 4; ++kt) a[kt] = load_a_f32(s1, m0, kt * 32, HH);
  const int col0  = (lane & 15);
  const int mbase = m0 + ((lane & 16) ? 8 : 0);
#pragma unroll
  for (int nt = 0; nt < 4; ++nt) {
    v8f c = {};
#pragma unroll
    for (int kt = 0; kt < 4; ++kt)
      c = wmma16(a[kt], load_bp(bpack, 4, kt, nt), c);
    const int col = nt * 16 + col0;
#pragma unroll
    for (int r = 0; r < 8; ++r)
      s[(size_t)(mbase + r) * 64 + col] = fmaxf(c[r], 0.0f);
  }
}

// ---------------------------------------------------------------------------
// Stage 6: heads (tiny, fp32 VALU for full precision). One block per graph.
// ---------------------------------------------------------------------------
__global__ __launch_bounds__(256) void k_heads(const float* __restrict__ s,
                                               const float* __restrict__ Wa,
                                               const float* __restrict__ Ws,
                                               const float* __restrict__ Wt,
                                               const float* __restrict__ Wact,
                                               const float* __restrict__ Wv1,
                                               const float* __restrict__ Wv2,
                                               float* __restrict__ out) {
  const int gph = blockIdx.x;
  const int tid = threadIdx.x;
  __shared__ float sr[64];
  __shared__ float t32[32];
  if (tid < 64) sr[tid] = s[(size_t)gph * 64 + tid];
  __syncthreads();
  if (tid < 32) {
    float acc = 0.0f;
#pragma unroll
    for (int k = 0; k < 64; ++k) acc += sr[k] * Wv1[k * 32 + tid];
    t32[tid] = fmaxf(acc, 0.0f);
  }
  __syncthreads();
  if (tid == 0) {
    float acc = 0.0f;
#pragma unroll
    for (int k = 0; k < 32; ++k) acc += t32[k] * Wv2[k];
    out[OUT_V0 + gph] = tanhf(acc);
  }
  for (int c = tid; c < NACT + 2 * MAXNEU + NACTV; c += blockDim.x) {
    const float* W;
    int col, ncols;
    size_t off;
    if (c < NACT)                 { W = Wa;   col = c;                ncols = NACT;   off = OUT_A0   + (size_t)gph * NACT   + col; }
    else if (c < NACT + MAXNEU)   { W = Ws;   col = c - NACT;         ncols = MAXNEU; off = OUT_S0   + (size_t)gph * MAXNEU + col; }
    else if (c < NACT + 2*MAXNEU) { W = Wt;   col = c - NACT - MAXNEU; ncols = MAXNEU; off = OUT_T0   + (size_t)gph * MAXNEU + col; }
    else                          { W = Wact; col = c - NACT - 2*MAXNEU; ncols = NACTV; off = OUT_ACT0 + (size_t)gph * NACTV  + col; }
    float acc = 0.0f;
#pragma unroll
    for (int k = 0; k < 64; ++k) acc += sr[k] * W[k * ncols + col];
    out[off] = acc;
  }
}

// ---------------------------------------------------------------------------
// Host launcher
// ---------------------------------------------------------------------------
extern "C" void kernel_launch(void* const* d_in, const int* in_sizes, int n_in,
                              void* d_out, int out_size, void* d_ws, size_t ws_size,
                              hipStream_t stream) {
  const float* nf    = (const float*)d_in[0];   // [1,N,32]
  const int*   ei    = (const int*)  d_in[1];   // [2,E]
  const float* ew    = (const float*)d_in[2];   // [E]
  // d_in[3] layer_positions: unused by reference
  const int*   batch = (const int*)  d_in[4];   // [N] sorted
  // d_in[5] num_graphs scalar
  const float* W_in  = (const float*)d_in[6];   // [32,128]
  const float* W_msg = (const float*)d_in[7];   // [128,128]
  const float* W1    = (const float*)d_in[8];   // [256,128]
  const float* W2    = (const float*)d_in[9];   // [128,64]
  const float* Wa    = (const float*)d_in[10];  // [64,5]
  const float* Ws    = (const float*)d_in[11];  // [64,1000]
  const float* Wt    = (const float*)d_in[12];  // [64,1000]
  const float* Wact  = (const float*)d_in[13];  // [64,4]
  const float* Wv1   = (const float*)d_in[14];  // [64,32]
  const float* Wv2   = (const float*)d_in[15];  // [32,1]
  float* out = (float*)d_out;

  // workspace layout (floats)
  float* ws = (float*)d_ws;
  const size_t NH = (size_t)NN * HH;            // 16,777,216
  float* h0   = ws;                              // [N,H]
  float* hbuf = h0   + NH;                       // msg, then h (in place)
  float* seg  = hbuf + NH;                       // [G,H]
  float* cnt  = seg  + (size_t)GG * HH;          // [G]
  float* gpo  = cnt  + GG;                       // [G,H]
  float* s1   = gpo  + (size_t)GG * HH;          // [G,128]
  float* sb   = s1   + (size_t)GG * HH;          // [G,64]
  _Float16* pWin  = (_Float16*)(sb + (size_t)GG * 64);
  _Float16* pWmsg = pWin  + 4096;                // 32x128 packed
  _Float16* pW1   = pWmsg + 16384;               // 128x128 packed
  _Float16* pW2   = pW1   + 32768;               // 256x128 packed
                                                 // pW2: 128x64 -> 8192 halves

  // 0) zero accumulators (atomics accumulate each call)
  k_zero<<<(int)((NH + 255) / 256), 256, 0, stream>>>(hbuf, (int)NH);
  k_zero<<<(GG * HH + 255) / 256, 256, 0, stream>>>(seg, GG * HH);
  k_zero<<<(GG + 255) / 256, 256, 0, stream>>>(cnt, GG);

  // 1) pack weights into WMMA-B lane layout (f16)
  k_pack_b<<<(4096  + 255) / 256, 256, 0, stream>>>(W_in,  pWin,  32,  128);
  k_pack_b<<<(16384 + 255) / 256, 256, 0, stream>>>(W_msg, pWmsg, 128, 128);
  k_pack_b<<<(32768 + 255) / 256, 256, 0, stream>>>(W1,    pW1,   256, 128);
  k_pack_b<<<(8192  + 255) / 256, 256, 0, stream>>>(W2,    pW2,   128, 64);

  // 2) h0 = X @ W_in  (WMMA)
  k_h0<<<NN / 128, 256, 0, stream>>>(nf, pWin, h0);

  // 3) edge-weighted message scatter
  k_edges<<<(int)(((size_t)EE * HH) / 256), 256, 0, stream>>>(ei, ew, h0, hbuf);

  // 4) h = relu(h0 + msg @ W_msg)  (WMMA, in-place over msg)
  k_msggemm<<<NN / 128, 256, 0, stream>>>(h0, hbuf, pWmsg);

  // 5) pooled mean per graph (sorted batch -> run-length + few atomics)
  k_cnt<<<NN / 256, 256, 0, stream>>>(batch, cnt);
  k_pool<<<NN / POOL_CHUNK, 128, 0, stream>>>(hbuf, batch, seg);
  k_mean<<<(GG * HH + 255) / 256, 256, 0, stream>>>(seg, cnt, gpo);

  // 6) backbone (WMMA): s1 = relu([g,g] @ W1); s = relu(s1 @ W2)
  k_s1<<<GG / 128, 256, 0, stream>>>(gpo, pW1, s1);
  k_s2<<<GG / 128, 256, 0, stream>>>(s1, pW2, sb);

  // 7) heads (fp32 VALU, tiny)
  k_heads<<<GG, 256, 0, stream>>>(sb, Wa, Ws, Wt, Wact, Wv1, Wv2, out);

  (void)in_sizes; (void)n_in; (void)out_size; (void)ws_size;
}
